// Regressor2_41944650613178
// MI455X (gfx1250) — compile-verified, matrix operand order checked by reading, not run on hardware
//
#include <hip/hip_runtime.h>
#include <hip/hip_bf16.h>
#include <math.h>

#define HH 448
#define WW 608
#define CCLS 128
#define INF_ 128
#define WT 32
#define HW (HH*WW)

typedef __attribute__((ext_vector_type(16))) __bf16 v16bf;
typedef __attribute__((ext_vector_type(8)))  float  v8f;
typedef __attribute__((ext_vector_type(4)))  unsigned int v4u;
typedef __attribute__((ext_vector_type(8)))  unsigned int v8u;

__device__ __forceinline__ unsigned short f2bf(float f) {
    unsigned u = __builtin_bit_cast(unsigned, f);
    u += 0x7FFFu + ((u >> 16) & 1u);            // round-to-nearest-even
    return (unsigned short)(u >> 16);
}
__device__ __forceinline__ float lrelu(float v) { return v > 0.f ? v : 0.01f * v; }

// Load one 16x32-layout WMMA fragment half-pair for this lane:
// 8 contiguous bf16 at p (K-chunk 1) and 8 contiguous bf16 at p+16 (K-chunk 2).
// Pure vector IR: 2x ds_load_b128 -> concat -> bitcast (no scalar chains).
__device__ __forceinline__ v16bf load_frag16(const unsigned short* p) {
    v4u lo = *(const v4u*)(p);
    v4u hi = *(const v4u*)(p + 16);
    v8u cat = __builtin_shufflevector(lo, hi, 0, 1, 2, 3, 4, 5, 6, 7);
    return __builtin_bit_cast(v16bf, cat);
}

// ---------------------------------------------------------------------------
// Kernel 1: per-row classifier GEMM (bf16 WMMA) + softmax + class loss
// One block per row h. 256 threads = 8 waves; wave j owns c-rows [16j,16j+16).
// ---------------------------------------------------------------------------
__global__ __launch_bounds__(256) void k_classify(
    const float* __restrict__ x, const float* __restrict__ x_gt,
    const float* __restrict__ Wc, const float* __restrict__ bc,
    float* __restrict__ out)
{
    __shared__ alignas(16) unsigned short sWc[CCLS][INF_ + 8];  // 34,816 B, bf16
    __shared__ alignas(16) unsigned short sXt[WT][INF_ + 8];    //  8,704 B, bf16, x transposed [w][k]
    __shared__ float sLog[CCLS][WT + 1];                        // 16,896 B
    __shared__ float sRed[8][WT];
    __shared__ float sMax[WT];
    __shared__ float sSum[WT];
    __shared__ float sPl[WT];

    const int h    = blockIdx.x;
    const int tid  = threadIdx.x;
    const int lane = tid & 31;
    const int jw   = tid >> 5;
    const int cbase = jw * 16;

    // Stage Wc[h] (128x128 f32 -> bf16). Contiguous global reads.
    const float* Wch = Wc + (size_t)h * CCLS * INF_;
    for (int e = tid; e < CCLS * INF_; e += 256) {
        int c = e >> 7, i = e & 127;
        sWc[c][i] = f2bf(Wch[e]);
    }
    const float* bch = bc + h * CCLS;

    for (int wb = 0; wb < WW; wb += WT) {
        __syncthreads();   // protect sXt/sLog/sRed reuse across tiles

        // Stage x tile transposed: sXt[w][k] = x[k,h,wb+w]  (coalesced over w)
        {
            int w = tid & 31;
            for (int k = tid >> 5; k < INF_; k += 8)
                sXt[w][k] = f2bf(x[((size_t)k * HH + h) * WW + wb + w]);
        }
        __syncthreads();

        // WMMA: D(16c x 16w) += A(16c x 32k bf16) * B(32k x 16w bf16)
        // ISA A layout: lanes 0-15 hold K = kb..kb+7 and kb+16..kb+23 of row M=lane;
        //               lanes 16-31 hold K = kb+8.. and kb+24.. ; B mirrored per column.
        v8f acc0 = {0.f,0.f,0.f,0.f,0.f,0.f,0.f,0.f};
        v8f acc1 = {0.f,0.f,0.f,0.f,0.f,0.f,0.f,0.f};
        const int crow  = cbase + (lane & 15);
        const int khalf = (lane < 16) ? 0 : 8;
        const int wc0   = lane & 15;        // B column, n-tile 0
        const int wc1   = 16 + (lane & 15); // B column, n-tile 1
        #pragma unroll
        for (int kt = 0; kt < 4; ++kt) {
            const int kb = kt * 32 + khalf;
            v16bf A  = load_frag16(&sWc[crow][kb]);
            v16bf B0 = load_frag16(&sXt[wc0][kb]);
            v16bf B1 = load_frag16(&sXt[wc1][kb]);
            acc0 = __builtin_amdgcn_wmma_f32_16x16x32_bf16(
                       false, A, false, B0, (short)0, acc0, false, false);
            acc1 = __builtin_amdgcn_wmma_f32_16x16x32_bf16(
                       false, A, false, B1, (short)0, acc1, false, false);
        }

        // bias + leaky-relu -> sLog  (D layout: VGPR r, lanes<16 -> M=r, lanes>=16 -> M=r+8)
        const int hi8 = (lane < 16) ? 0 : 8;
        #pragma unroll
        for (int r = 0; r < 8; ++r) {
            const int c = cbase + hi8 + r;
            const float bias = bch[c];
            sLog[c][lane & 15]        = lrelu(acc0[r] + bias);
            sLog[c][16 + (lane & 15)] = lrelu(acc1[r] + bias);
        }
        __syncthreads();

        // Softmax over c (128 values per column), then loss = log(sum exp(p)) - p_label
        const int wl = tid & 31;
        const int cs = tid >> 5;           // 8 segments of 16 classes
        float m = -1e30f;
        #pragma unroll
        for (int cc = 0; cc < 16; ++cc) m = fmaxf(m, sLog[cs * 16 + cc][wl]);
        sRed[cs][wl] = m;
        __syncthreads();
        if (cs == 0) {
            float mm = sRed[0][wl];
            for (int s2 = 1; s2 < 8; ++s2) mm = fmaxf(mm, sRed[s2][wl]);
            sMax[wl] = mm;
        }
        __syncthreads();
        const float colmax = sMax[wl];
        float se = 0.f;
        #pragma unroll
        for (int cc = 0; cc < 16; ++cc) se += __expf(sLog[cs * 16 + cc][wl] - colmax);
        sRed[cs][wl] = se;
        __syncthreads();
        if (cs == 0) {
            float ss = 0.f;
            for (int s2 = 0; s2 < 8; ++s2) ss += sRed[s2][wl];
            sSum[wl] = ss;
        }
        __syncthreads();
        const float invs = 1.f / sSum[wl];
        float g = x_gt[h * WW + wb + wl];
        int label = (int)(g * 128.f);
        label = label < 0 ? 0 : (label > 127 ? 127 : label);
        float z = 0.f;
        #pragma unroll
        for (int cc = 0; cc < 16; ++cc) {
            const int c = cs * 16 + cc;
            float p = __expf(sLog[c][wl] - colmax) * invs;
            z += __expf(p);
            if (c == label) sPl[wl] = p;
        }
        sRed[cs][wl] = z;
        __syncthreads();
        if (cs == 0) {
            float zz = 0.f;
            for (int s2 = 0; s2 < 8; ++s2) zz += sRed[s2][wl];
            out[2 * HW + h * WW + wb + wl] = logf(zz) - sPl[wl];
        }
    }
}

// ---------------------------------------------------------------------------
// Kernel 2: per-pixel gathered-expert MLP 128->32->32->2.
// One block per row; one wave per pixel; lane = output neuron index.
// Weight loads are 128 B contiguous per lane-group and stream sequentially
// through each 16 KB expert block (L2-resident per row).
// ---------------------------------------------------------------------------
__global__ __launch_bounds__(256) void k_mlp(
    const float* __restrict__ x, const float* __restrict__ x_gt,
    const float* __restrict__ W1, const float* __restrict__ b1,
    const float* __restrict__ W2, const float* __restrict__ b2,
    const float* __restrict__ Wrc, const float* __restrict__ brc,
    float* __restrict__ out)
{
    __shared__ float sXf[8][INF_];   // 4 KB: 8 pixels x 128 features
    const int h    = blockIdx.x;
    const int tid  = threadIdx.x;
    const int lane = tid & 31;
    const int jw   = tid >> 5;

    for (int wb = 0; wb < WW; wb += 8) {
        __syncthreads();
        // Cooperative stage of 8 pixels' features (coalesced 32 B runs)
        for (int e = tid; e < 8 * INF_; e += 256) {
            int p = e & 7, k = e >> 3;
            sXf[p][k] = x[((size_t)k * HH + h) * WW + wb + p];
        }
        __syncthreads();

        const int w = wb + jw;
        float g = x_gt[h * WW + w];
        int ind = (int)(g * 128.f);
        ind = ind < 0 ? 0 : (ind > 127 ? 127 : ind);
        const size_t fe = (size_t)h * 128 + ind;

        // Layer 1: 128 -> 32
        const float* W1p = W1 + fe * (size_t)(INF_ * 32);
        float a1 = b1[fe * 32 + lane];
        #pragma unroll 8
        for (int i = 0; i < INF_; ++i)
            a1 = fmaf(sXf[jw][i], W1p[i * 32 + lane], a1);
        a1 = lrelu(a1);

        // Layer 2: 32 -> 32 (broadcast via v_readlane)
        const float* W2p = W2 + fe * (size_t)(32 * 32);
        float a2 = b2[fe * 32 + lane];
        #pragma unroll
        for (int j = 0; j < 32; ++j)
            a2 = fmaf(__shfl(a1, j, 32), W2p[j * 32 + lane], a2);
        a2 = lrelu(a2);

        // Layer 3: 32 -> 2 (lane-parallel partials, xor-shuffle reduce)
        const float* Wrcp = Wrc + fe * 64;
        float p0 = a2 * Wrcp[lane * 2 + 0];
        float p1 = a2 * Wrcp[lane * 2 + 1];
        #pragma unroll
        for (int off = 16; off > 0; off >>= 1) {
            p0 += __shfl_xor(p0, off, 32);
            p1 += __shfl_xor(p1, off, 32);
        }
        float rg = lrelu(p0 + brc[fe * 2 + 0]);
        float mk = lrelu(p1 + brc[fe * 2 + 1]);
        if (lane == 0) {
            out[h * WW + w]      = (float)ind * (1.f / 128.f) + rg * (1.f / 128.f);
            out[HW + h * WW + w] = mk;
        }
    }
}

extern "C" void kernel_launch(void* const* d_in, const int* in_sizes, int n_in,
                              void* d_out, int out_size, void* d_ws, size_t ws_size,
                              hipStream_t stream)
{
    const float* x    = (const float*)d_in[0];
    const float* x_gt = (const float*)d_in[1];
    const float* Wc   = (const float*)d_in[2];
    const float* bc   = (const float*)d_in[3];
    const float* W1   = (const float*)d_in[4];
    const float* b1   = (const float*)d_in[5];
    const float* W2   = (const float*)d_in[6];
    const float* b2   = (const float*)d_in[7];
    const float* Wrc  = (const float*)d_in[8];
    const float* brc  = (const float*)d_in[9];
    float* out = (float*)d_out;

    hipLaunchKernelGGL(k_classify, dim3(HH), dim3(256), 0, stream,
                       x, x_gt, Wc, bc, out);
    hipLaunchKernelGGL(k_mlp, dim3(HH), dim3(256), 0, stream,
                       x, x_gt, W1, b1, W2, b2, Wrc, brc, out);
}
